// TrustSAGE_1812476199451
// MI455X (gfx1250) — matrix operand order repaired and checked by reading, not compiled, not run on hardware
//
#include <hip/hip_runtime.h>
#include <math.h>

#define NNODES 50000
#define NEDGES 800000
#define INCH   128
#define HIDN   64
#define NHEAD  4
#define NTRIP  200000
#define ETOT   (NEDGES + NNODES)
#define NEGSL  0.2f
#define EPSV   1e-5f

typedef __attribute__((ext_vector_type(16))) _Float16 v16h;
typedef __attribute__((ext_vector_type(8)))  _Float16 v8h;
typedef __attribute__((ext_vector_type(8)))  float    v8f;

// ---------------------------------------------------------------------------
// fp32 -> f16 elementwise convert (A operand staging)
// ---------------------------------------------------------------------------
__global__ void f32_to_f16(const float* __restrict__ in,
                           _Float16* __restrict__ out, long long n) {
    long long i = (long long)blockIdx.x * blockDim.x + threadIdx.x;
    if (i < n) out[i] = (_Float16)in[i];
}

// fp32 W[K,N] -> f16 WT[N,K] (transposed so B-fragment loads are contiguous)
__global__ void w_to_f16T(const float* __restrict__ W,
                          _Float16* __restrict__ WT, int K, int N) {
    int idx = blockIdx.x * blockDim.x + threadIdx.x;   // n*K + k
    if (idx >= K * N) return;
    int nn = idx / K, k = idx - nn * K;
    WT[idx] = (_Float16)W[(size_t)k * N + nn];
}

// ---------------------------------------------------------------------------
// WMMA GEMM: C[M,N] = A[M,K] @ B[K,N], A f16 [M,K], B given as WT f16 [N,K].
// One wave32 computes a 16(M) x 64(N) strip = 4 WMMA tiles sharing one A
// fragment per K-step. All fragment loads are 16B global_load_b128, no
// converts in the hot loop. K mult of 32, M mult of 16, N mult of 64.
// ---------------------------------------------------------------------------
__global__ __launch_bounds__(256)
void gemm_wmma_f16(const _Float16* __restrict__ A,
                   const _Float16* __restrict__ WT,
                   float* __restrict__ C, int M, int K, int N) {
    const int wave    = (int)((blockIdx.x * blockDim.x + threadIdx.x) >> 5);
    const int lane    = threadIdx.x & 31;
    const int stripsN = N >> 6;                 // 64-wide N strips
    const int tilesM  = M >> 4;
    const int tm = wave / stripsN;
    const int tn = wave % stripsN;
    if (tm >= tilesM) return;                   // uniform per-wave exit

    const int r16 = lane & 15;                  // row (A) / col (B,C,D)
    const int kh  = lane >> 4;                  // K-half selector

    v8f acc[4] = {{}, {}, {}, {}};
    const _Float16* Arow = A + (size_t)(tm * 16 + r16) * K;

    for (int k0 = 0; k0 < K; k0 += 32) {
        // prefetch next K slab of A (speculative; dropped if OOB)
        __builtin_prefetch(Arow + k0 + 64, 0, 0);

        // A fragment (16x32 f16): halves 0-7 -> K = k0+kh*8+j,
        // halves 8-15 -> K = k0+16+kh*8+(j-8); two contiguous 16B runs.
        v8h alo = *(const v8h*)(Arow + k0 + kh * 8);
        v8h ahi = *(const v8h*)(Arow + k0 + 16 + kh * 8);
        v16h a = __builtin_shufflevector(alo, ahi,
                 0, 1, 2, 3, 4, 5, 6, 7, 8, 9, 10, 11, 12, 13, 14, 15);

#pragma unroll
        for (int s = 0; s < 4; ++s) {
            // B fragment (32x16 f16): halves 0-7 -> K = k0+kh*8+j,
            // halves 8-15 -> K = k0+16+kh*8+(j-8). Col = lane&15.
            // WT[N,K]: two 8-half runs 16 apart -> two b128 loads.
            const int col = tn * 64 + s * 16 + r16;
            const _Float16* Bp = WT + (size_t)col * K + k0 + kh * 8;
            v8h blo = *(const v8h*)(Bp);
            v8h bhi = *(const v8h*)(Bp + 16);
            v16h b = __builtin_shufflevector(blo, bhi,
                     0, 1, 2, 3, 4, 5, 6, 7, 8, 9, 10, 11, 12, 13, 14, 15);
            acc[s] = __builtin_amdgcn_wmma_f32_16x16x32_f16(
                false, a, false, b, (short)0, acc[s], false, false);
        }
    }
    // C/D layout: VGPR r -> M = r + 8*(lane>=16), N = lane&15
#pragma unroll
    for (int s = 0; s < 4; ++s) {
#pragma unroll
        for (int r = 0; r < 8; ++r) {
            int m = tm * 16 + r + kh * 8;
            C[(size_t)m * N + (tn * 64 + s * 16 + r16)] = acc[s][r];
        }
    }
}

// ---------------------------------------------------------------------------
// per-(node,head) attention coefficients: as = <h, a_src>, ad = <h, a_dst>
// ---------------------------------------------------------------------------
__global__ void attn_coeff(const float* __restrict__ h,
                           const float* __restrict__ a_src,
                           const float* __restrict__ a_dst,
                           float* __restrict__ as_, float* __restrict__ ad_,
                           int n) {
    int idx = blockIdx.x * blockDim.x + threadIdx.x;   // node*4 + head
    if (idx >= n * NHEAD) return;
    int node = idx >> 2, head = idx & 3;
    const float* hp = h + (size_t)node * (NHEAD * HIDN) + head * HIDN;
    const float* sp = a_src + head * HIDN;
    const float* dp = a_dst + head * HIDN;
    float s = 0.f, d = 0.f;
#pragma unroll 8
    for (int c = 0; c < HIDN; ++c) { s += hp[c] * sp[c]; d += hp[c] * dp[c]; }
    as_[idx] = s; ad_[idx] = d;
}

// order-preserving uint encoding for float atomicMax
__device__ __forceinline__ unsigned fenc(float f) {
    unsigned u = __float_as_uint(f);
    return (u & 0x80000000u) ? ~u : (u | 0x80000000u);
}
__device__ __forceinline__ float fdec(unsigned u) {
    unsigned b = (u & 0x80000000u) ? (u ^ 0x80000000u) : ~u;
    return __uint_as_float(b);
}

__device__ __forceinline__ void edge_sd(const int* __restrict__ ei, int e,
                                        int& s, int& d) {
    if (e < NEDGES) { s = ei[e]; d = ei[NEDGES + e]; }
    else            { s = d = e - NEDGES; }            // appended self-loops
}

__global__ void fill_u32(unsigned* __restrict__ p, unsigned v, long long n) {
    long long i = (long long)blockIdx.x * blockDim.x + threadIdx.x;
    if (i < n) p[i] = v;
}

// pass 1: scatter-max of leaky_relu(as[src]+ad[dst]) over dst
__global__ void edge_max(const int* __restrict__ ei,
                         const float* __restrict__ as_,
                         const float* __restrict__ ad_,
                         unsigned* __restrict__ menc) {
    int idx = blockIdx.x * blockDim.x + threadIdx.x;   // edge*4 + head
    if (idx >= ETOT * NHEAD) return;
    int e = idx >> 2, hh = idx & 3;
    int s, d; edge_sd(ei, e, s, d);
    float x = as_[s * NHEAD + hh] + ad_[d * NHEAD + hh];
    x = (x > 0.f) ? x : NEGSL * x;
    atomicMax(&menc[d * NHEAD + hh], fenc(x));
}

// pass 2: p = exp(e - max[dst]); scatter-add into s[dst]
__global__ void edge_expsum(const int* __restrict__ ei,
                            const float* __restrict__ as_,
                            const float* __restrict__ ad_,
                            const unsigned* __restrict__ menc,
                            float* __restrict__ pbuf,
                            float* __restrict__ ssum) {
    int idx = blockIdx.x * blockDim.x + threadIdx.x;
    if (idx >= ETOT * NHEAD) return;
    int e = idx >> 2, hh = idx & 3;
    int s, d; edge_sd(ei, e, s, d);
    float x = as_[s * NHEAD + hh] + ad_[d * NHEAD + hh];
    x = (x > 0.f) ? x : NEGSL * x;
    float pe = __expf(x - fdec(menc[d * NHEAD + hh]));
    pbuf[idx] = pe;
    atomicAdd(&ssum[d * NHEAD + hh], pe);
}

// pass 3: agg[dst] += alpha * h[src]; 64 channel-threads per edge, heads looped.
// Whole h/agg working set (~102 MB) is resident in the 192 MB L2, so the
// scatter atomics resolve in L2.
__global__ void edge_agg(const int* __restrict__ ei,
                         const float* __restrict__ h,
                         const float* __restrict__ pbuf,
                         const float* __restrict__ ssum,
                         float* __restrict__ agg) {
    long long gid = (long long)blockIdx.x * blockDim.x + threadIdx.x;
    int e = (int)(gid >> 6);
    int c = (int)(gid & 63);
    if (e >= ETOT) return;
    int s, d; edge_sd(ei, e, s, d);
#pragma unroll
    for (int hh = 0; hh < NHEAD; ++hh) {
        float alpha = pbuf[e * NHEAD + hh] / (ssum[d * NHEAD + hh] + 1e-16f);
        atomicAdd(&agg[(size_t)d * (NHEAD * HIDN) + hh * HIDN + c],
                  h[(size_t)s * (NHEAD * HIDN) + hh * HIDN + c] * alpha);
    }
}

// mean over heads + bias -> node features [N, HIDN]
__global__ void head_mean(const float* __restrict__ agg,
                          const float* __restrict__ bias,
                          float* __restrict__ node, int n) {
    int gid = blockIdx.x * blockDim.x + threadIdx.x;
    if (gid >= n * HIDN) return;
    int i = gid >> 6, c = gid & 63;
    const float* ap = agg + (size_t)i * (NHEAD * HIDN);
    float v = 0.25f * (ap[c] + ap[HIDN + c] + ap[2 * HIDN + c] + ap[3 * HIDN + c]);
    node[gid] = v + bias[c];
}

// GraphNorm reductions: per-block partial column sums, 64 channels = blockDim
__global__ void col_sum(const float* __restrict__ x, float* __restrict__ sums,
                        int n, int npb) {
    int c = threadIdx.x;
    int s0 = blockIdx.x * npb;
    int s1 = min(s0 + npb, n);
    float s = 0.f;
    for (int i = s0; i < s1; ++i) s += x[(size_t)i * HIDN + c];
    atomicAdd(&sums[c], s);
}

__global__ void col_sq(const float* __restrict__ x,
                       const float* __restrict__ sums,
                       const float* __restrict__ mscale,
                       float* __restrict__ sq, int n, int npb) {
    int c = threadIdx.x;
    float shift = (sums[c] / (float)n) * mscale[c];
    int s0 = blockIdx.x * npb;
    int s1 = min(s0 + npb, n);
    float q = 0.f;
    for (int i = s0; i < s1; ++i) {
        float d = x[(size_t)i * HIDN + c] - shift;
        q += d * d;
    }
    atomicAdd(&sq[c], q);
}

__global__ void gnorm_relu(float* __restrict__ x,
                           const float* __restrict__ sums,
                           const float* __restrict__ sq,
                           const float* __restrict__ w,
                           const float* __restrict__ b,
                           const float* __restrict__ mscale, int n) {
    int gid = blockIdx.x * blockDim.x + threadIdx.x;
    if (gid >= n * HIDN) return;
    int c = gid & 63;
    float shift = (sums[c] / (float)n) * mscale[c];
    float var   = sq[c] / (float)n;
    float y = w[c] * (x[gid] - shift) * rsqrtf(var + EPSV) + b[c];
    x[gid] = fmaxf(y, 0.f);
}

// triplet head: one wave32 per triplet; gather 4x64, layernorm-256, dot fc_w
__global__ __launch_bounds__(256)
void triplet_head(const float* __restrict__ emb, const int* __restrict__ trip,
                  const float* __restrict__ lnw, const float* __restrict__ lnb,
                  const float* __restrict__ fcw, const float* __restrict__ fcb,
                  float* __restrict__ out, int T) {
    int warp = (int)((blockIdx.x * blockDim.x + threadIdx.x) >> 5);
    int lane = threadIdx.x & 31;
    if (warp >= T) return;
    const int* tr = trip + (size_t)warp * 4;
    float v[8];
    float s = 0.f;
#pragma unroll
    for (int j = 0; j < 8; ++j) {
        int f = lane + 32 * j;                    // 0..255
        int node = tr[f >> 6];
        v[j] = emb[(size_t)node * HIDN + (f & 63)];
        s += v[j];
    }
#pragma unroll
    for (int off = 16; off; off >>= 1) s += __shfl_xor(s, off, 32);
    float mu = s * (1.f / 256.f);
    float q = 0.f;
#pragma unroll
    for (int j = 0; j < 8; ++j) { float d = v[j] - mu; q += d * d; }
#pragma unroll
    for (int off = 16; off; off >>= 1) q += __shfl_xor(q, off, 32);
    float rstd = rsqrtf(q * (1.f / 256.f) + EPSV);
    float acc = 0.f;
#pragma unroll
    for (int j = 0; j < 8; ++j) {
        int f = lane + 32 * j;
        acc += ((v[j] - mu) * rstd * lnw[f] + lnb[f]) * fcw[f];
    }
#pragma unroll
    for (int off = 16; off; off >>= 1) acc += __shfl_xor(acc, off, 32);
    if (lane == 0) out[warp] = acc + fcb[0];
}

// ---------------------------------------------------------------------------
// host side
// ---------------------------------------------------------------------------
static inline int cdiv(long long a, int b) { return (int)((a + b - 1) / b); }

static void run_gat_layer(const float* in_feat, int K, const float* W,
                          const float* a_s, const float* a_d, const float* bias,
                          const float* gw, const float* gb, const float* gs,
                          const int* ei,
                          _Float16* a16, _Float16* wt16,
                          float* h_big, float* agg, float* node,
                          float* as_, float* ad_, unsigned* menc, float* ssum,
                          float* pbuf, float* csum, float* csq,
                          hipStream_t stream) {
    const int Nout = NHEAD * HIDN;                       // 256

    // stage operands in f16 (A as-is, W transposed to [N,K])
    f32_to_f16<<<cdiv((long long)NNODES * K, 256), 256, 0, stream>>>(
        in_feat, a16, (long long)NNODES * K);
    w_to_f16T<<<cdiv((long long)K * Nout, 256), 256, 0, stream>>>(
        W, wt16, K, Nout);

    // GEMM: one wave per 16x64 strip -> waves = (M/16)*(N/64)
    long long waves = (long long)(NNODES / 16) * (Nout / 64);
    gemm_wmma_f16<<<cdiv(waves * 32, 256), 256, 0, stream>>>(
        a16, wt16, h_big, NNODES, K, Nout);

    attn_coeff<<<cdiv((long long)NNODES * NHEAD, 256), 256, 0, stream>>>(
        h_big, a_s, a_d, as_, ad_, NNODES);

    fill_u32<<<cdiv((long long)NNODES * NHEAD, 256), 256, 0, stream>>>(
        menc, 0x007FFFFFu, (long long)NNODES * NHEAD);    // enc(-inf)
    fill_u32<<<cdiv((long long)NNODES * NHEAD, 256), 256, 0, stream>>>(
        (unsigned*)ssum, 0u, (long long)NNODES * NHEAD);
    fill_u32<<<cdiv((long long)NNODES * Nout, 256), 256, 0, stream>>>(
        (unsigned*)agg, 0u, (long long)NNODES * Nout);
    fill_u32<<<1, 128, 0, stream>>>((unsigned*)csum, 0u, HIDN);
    fill_u32<<<1, 128, 0, stream>>>((unsigned*)csq, 0u, HIDN);

    edge_max<<<cdiv((long long)ETOT * NHEAD, 256), 256, 0, stream>>>(
        ei, as_, ad_, menc);
    edge_expsum<<<cdiv((long long)ETOT * NHEAD, 256), 256, 0, stream>>>(
        ei, as_, ad_, menc, pbuf, ssum);
    edge_agg<<<cdiv((long long)ETOT * 64, 256), 256, 0, stream>>>(
        ei, h_big, pbuf, ssum, agg);
    head_mean<<<cdiv((long long)NNODES * HIDN, 256), 256, 0, stream>>>(
        agg, bias, node, NNODES);

    const int NPB = 256;
    col_sum<<<cdiv(NNODES, NPB), HIDN, 0, stream>>>(node, csum, NNODES, NPB);
    col_sq <<<cdiv(NNODES, NPB), HIDN, 0, stream>>>(node, csum, gs, csq, NNODES, NPB);
    gnorm_relu<<<cdiv((long long)NNODES * HIDN, 256), 256, 0, stream>>>(
        node, csum, csq, gw, gb, gs, NNODES);
}

extern "C" void kernel_launch(void* const* d_in, const int* in_sizes, int n_in,
                              void* d_out, int out_size, void* d_ws, size_t ws_size,
                              hipStream_t stream) {
    (void)in_sizes; (void)n_in; (void)out_size; (void)ws_size;
    const float* x       = (const float*)d_in[0];
    // d_in[1] = edge_weight (unused by reference)
    const float* W1      = (const float*)d_in[2];
    const float* a_src1  = (const float*)d_in[3];
    const float* a_dst1  = (const float*)d_in[4];
    const float* b1      = (const float*)d_in[5];
    const float* W2      = (const float*)d_in[6];
    const float* a_src2  = (const float*)d_in[7];
    const float* a_dst2  = (const float*)d_in[8];
    const float* b2      = (const float*)d_in[9];
    const float* gn1_w   = (const float*)d_in[10];
    const float* gn1_b   = (const float*)d_in[11];
    const float* gn1_s   = (const float*)d_in[12];
    const float* gn2_w   = (const float*)d_in[13];
    const float* gn2_b   = (const float*)d_in[14];
    const float* gn2_s   = (const float*)d_in[15];
    const float* ln_w    = (const float*)d_in[16];
    const float* ln_b    = (const float*)d_in[17];
    const float* fc_w    = (const float*)d_in[18];
    const float* fc_b    = (const float*)d_in[19];
    const int*   ei      = (const int*)d_in[20];
    const int*   trip    = (const int*)d_in[21];
    float*       out     = (float*)d_out;

    // workspace layout (256B aligned)
    char* base = (char*)d_ws;
    size_t off = 0;
    auto alloc = [&](size_t bytes) {
        void* p = base + off;
        off += (bytes + 255) & ~(size_t)255;
        return p;
    };
    float*    h_big = (float*)alloc((size_t)NNODES * 256 * 4);     // 51.2 MB
    float*    agg   = (float*)alloc((size_t)NNODES * 256 * 4);     // 51.2 MB
    float*    node  = (float*)alloc((size_t)NNODES * 64 * 4);      // 12.8 MB
    float*    pbuf  = (float*)alloc((size_t)ETOT * NHEAD * 4);     // 13.6 MB
    _Float16* a16   = (_Float16*)alloc((size_t)NNODES * INCH * 2); // 12.8 MB
    _Float16* wt16  = (_Float16*)alloc((size_t)256 * INCH * 2);    // 64 KB
    float*    as_   = (float*)alloc((size_t)NNODES * NHEAD * 4);
    float*    ad_   = (float*)alloc((size_t)NNODES * NHEAD * 4);
    unsigned* menc  = (unsigned*)alloc((size_t)NNODES * NHEAD * 4);
    float*    ssum  = (float*)alloc((size_t)NNODES * NHEAD * 4);
    float*    csum  = (float*)alloc(HIDN * 4);
    float*    csq   = (float*)alloc(HIDN * 4);

    // layer 1: input x [N,128]
    run_gat_layer(x, INCH, W1, a_src1, a_dst1, b1, gn1_w, gn1_b, gn1_s, ei,
                  a16, wt16, h_big, agg, node, as_, ad_, menc, ssum, pbuf,
                  csum, csq, stream);
    // layer 2: input node [N,64] (normalized+relu'd in place)
    run_gat_layer(node, HIDN, W2, a_src2, a_dst2, b2, gn2_w, gn2_b, gn2_s, ei,
                  a16, wt16, h_big, agg, node, as_, ad_, menc, ssum, pbuf,
                  csum, csq, stream);

    // triplet scoring head
    triplet_head<<<cdiv((long long)NTRIP * 32, 256), 256, 0, stream>>>(
        node, trip, ln_w, ln_b, fc_w, fc_b, out, NTRIP);
}
